// MultiHeadAttention_34385508172252
// MI455X (gfx1250) — matrix operand order repaired and checked by reading, not compiled, not run on hardware
//
#include <hip/hip_runtime.h>

// MI455X / gfx1250 multi-head attention with fused projections.
// bf16 WMMA (v_wmma_f32_16x16x32_bf16) + async global->LDS staging (ASYNCcnt).
// Workspace: Q bf16 [B,NH,S,HD] | K bf16 [B,NH,S,HD] | V^T bf16 [B,NH,HD,S] = 48MB.

typedef __attribute__((ext_vector_type(16))) __bf16 v16bf;
typedef __attribute__((ext_vector_type(8)))  float  v8f;
typedef __attribute__((__vector_size__(16))) int    vi4;   // async-copy builtin operand type

#define B_   8
#define S_   1024
#define HID_ 1024
#define NH_  16
#define HD_  64

#define HAVE_ASYNC_LDS __has_builtin(__builtin_amdgcn_global_load_async_to_lds_b128)

__device__ __forceinline__ __bf16 to_bf16(float f) { return (__bf16)f; }

// 16-byte global -> LDS copy. Prefer the CDNA5 async path (GLOBAL_LOAD_ASYNC_TO_LDS_B128).
__device__ __forceinline__ void cp16_g2l(void* l, const void* g) {
#if HAVE_ASYNC_LDS
    __builtin_amdgcn_global_load_async_to_lds_b128((vi4*)g, (vi4*)l, 0, 0);
#else
    *(float4*)l = *(const float4*)g;           // fallback: load + ds_store
#endif
}

__device__ __forceinline__ void wait_async_copies() {
#if HAVE_ASYNC_LDS
#if __has_builtin(__builtin_amdgcn_s_wait_asynccnt)
    __builtin_amdgcn_s_wait_asynccnt(0);
#else
    asm volatile("s_wait_asynccnt 0" ::: "memory");
#endif
#endif
}

// ---------------------------------------------------------------------------
// Kernel 1: projections, 2x2 register-blocked (each wave computes a 32x32 tile).
// which = blockIdx.z: 0 -> Q = query@Wk^T+bk (x 1/sqrt(HD)), 1 -> K = key@Wk^T+bk,
// 2 -> V = value@Wv^T+bv stored transposed [b,h,d,s].
// ---------------------------------------------------------------------------
__global__ __launch_bounds__(256)
void proj_kernel(const float* __restrict__ Xq, const float* __restrict__ Xk,
                 const float* __restrict__ Xv,
                 const float* __restrict__ Wk, const float* __restrict__ bk,
                 const float* __restrict__ Wv, const float* __restrict__ bv,
                 __bf16* __restrict__ Qws, __bf16* __restrict__ Kws,
                 __bf16* __restrict__ Vtws)
{
    const int which = blockIdx.z;
    const float* X;  const float* W;  const float* bias;
    if (which == 0)      { X = Xq; W = Wk; bias = bk; }
    else if (which == 1) { X = Xk; W = Wk; bias = bk; }
    else                 { X = Xv; W = Wv; bias = bv; }

    const int lane = threadIdx.x & 31;
    const int wave = threadIdx.x >> 5;
    const int hi   = lane >> 4;
    const int l15  = lane & 15;

    const int tIdx = blockIdx.x * 8 + wave;    // 8192 tiles of 32x32
    const int m0   = (tIdx >> 5) * 32;         // token-row tile
    const int n0   = (tIdx & 31) * 32;         // hidden-col tile

    const int kb = hi * 8;                     // A kbase
    const int kh = hi * 8;                     // B kbase

    v8f acc00 = {}, acc01 = {}, acc10 = {}, acc11 = {};
    for (int k0 = 0; k0 < HID_; k0 += 32) {
        v16bf a0, a1, b0, b1;
        const float* ap0 = X + (size_t)(m0 + l15)      * HID_ + k0 + kb;
        const float* ap1 = X + (size_t)(m0 + 16 + l15) * HID_ + k0 + kb;
#pragma unroll
        for (int j = 0; j < 8; ++j) {
            a0[j] = to_bf16(ap0[j]); a0[8 + j] = to_bf16(ap0[16 + j]);
            a1[j] = to_bf16(ap1[j]); a1[8 + j] = to_bf16(ap1[16 + j]);
        }
        const float* bp0 = W + (size_t)(n0 + l15)      * HID_ + k0 + kh;
        const float* bp1 = W + (size_t)(n0 + 16 + l15) * HID_ + k0 + kh;
#pragma unroll
        for (int j = 0; j < 8; ++j) { b0[j] = to_bf16(bp0[j]); b0[8 + j] = to_bf16(bp0[16 + j]); b1[j] = to_bf16(bp1[j]); b1[8 + j] = to_bf16(bp1[16 + j]); }

        acc00 = __builtin_amdgcn_wmma_f32_16x16x32_bf16(false, a0, false, b0, (short)0, acc00, false, false);
        acc01 = __builtin_amdgcn_wmma_f32_16x16x32_bf16(false, a0, false, b1, (short)0, acc01, false, false);
        acc10 = __builtin_amdgcn_wmma_f32_16x16x32_bf16(false, a1, false, b0, (short)0, acc10, false, false);
        acc11 = __builtin_amdgcn_wmma_f32_16x16x32_bf16(false, a1, false, b1, (short)0, acc11, false, false);
    }

    const float scl = (which == 0) ? 0.125f : 1.0f;   // fold 1/sqrt(64) into Q
    const v8f* accs[2][2] = {{&acc00, &acc01}, {&acc10, &acc11}};
#pragma unroll
    for (int mi = 0; mi < 2; ++mi) {
#pragma unroll
        for (int ni = 0; ni < 2; ++ni) {
            const int   bcol = n0 + ni * 16 + l15;
            const float bn   = bias[bcol];
            const int   h    = bcol >> 6, d = bcol & (HD_ - 1);
            const v8f&  a    = *accs[mi][ni];
#pragma unroll
            for (int r = 0; r < 8; ++r) {
                const int m    = m0 + mi * 16 + r + 8 * hi;
                const int bidx = m >> 10, s = m & (S_ - 1);
                const __bf16 v = to_bf16((a[r] + bn) * scl);
                if (which == 0)      Qws [(((size_t)bidx * NH_ + h) * S_  + s) * HD_ + d] = v;
                else if (which == 1) Kws [(((size_t)bidx * NH_ + h) * S_  + s) * HD_ + d] = v;
                else                 Vtws[(((size_t)bidx * NH_ + h) * HD_ + d) * S_  + s] = v;
            }
        }
    }
}

// ---------------------------------------------------------------------------
// Kernel 2: flash attention. gridDim = (S/128, B*NH); 8 waves/block, each wave
// owns 16 query rows. K/V tiles staged once per block into LDS (async copies),
// shared by all 8 waves; 32 keys per step with online softmax.
// ---------------------------------------------------------------------------
__global__ __launch_bounds__(256)
void attn_kernel(const __bf16* __restrict__ Qws, const __bf16* __restrict__ Kws,
                 const __bf16* __restrict__ Vtws, const int* __restrict__ amask,
                 float* __restrict__ out)
{
    __shared__ __align__(16) __bf16 Kt[32][64];     // [key][d]   4KB
    __shared__ __align__(16) __bf16 Vtile[64][32];  // [d][key]   4KB
    __shared__ __bf16 pT[8][16][32];                // per-wave P transpose bounce, 8KB

    const int tid  = threadIdx.x;
    const int lane = tid & 31;
    const int wave = tid >> 5;
    const int hi   = lane >> 4;
    const int l15  = lane & 15;

    const int bh   = blockIdx.y;
    const int bidx = bh >> 4;
    const int h    = bh & 15;
    const int q0   = blockIdx.x * 128 + wave * 16;

    const __bf16* Qh = Qws  + (size_t)bh * S_ * HD_;
    const __bf16* Kh = Kws  + (size_t)bh * S_ * HD_;
    const __bf16* Vt = Vtws + (size_t)bh * HD_ * S_;
    const int*  mrow = amask + (size_t)bidx * S_;

    // Q A-fragments: two head-dim chunks of 32
    v16bf qa[2];
    {
        const int qrow = q0 + l15;
        const int kb   = hi * 8;
#pragma unroll
        for (int c = 0; c < 2; ++c) {
            const __bf16* qp = Qh + (size_t)qrow * HD_ + c * 32 + kb;
#pragma unroll
            for (int j = 0; j < 8; ++j) { qa[c][j] = qp[j]; qa[c][8 + j] = qp[16 + j]; }
        }
    }

    float mOld[8], lSum[8];
    v8f acc[4] = {{}, {}, {}, {}};
#pragma unroll
    for (int r = 0; r < 8; ++r) { mOld[r] = -3.0e38f; lSum[r] = 0.0f; }

    for (int key0 = 0; key0 < S_; key0 += 32) {
        // ---- stage K (contiguous 4KB) and V^T (64 rows x 64B) tiles into LDS ----
        cp16_g2l(&Kt[0][0] + tid * 8, Kh + (size_t)key0 * HD_ + tid * 8);
        {
            const int row = tid >> 2, ch = tid & 3;
            cp16_g2l(&Vtile[row][ch * 8], Vt + (size_t)row * S_ + key0 + ch * 8);
        }
        wait_async_copies();
        __syncthreads();

        // ---- scores: two 16-col tiles, K-dim 64 = 2 chained WMMAs each ----
        v8f sc[2];
#pragma unroll
        for (int t = 0; t < 2; ++t) {
            v16bf b0, b1;
            const __bf16* kp = &Kt[t * 16 + l15][hi * 8];
#pragma unroll
            for (int j = 0; j < 8; ++j) { b0[j] = kp[j]; b0[8 + j] = kp[16 + j]; b1[j] = kp[32 + j]; b1[8 + j] = kp[48 + j]; }
            v8f z = {};
            z = __builtin_amdgcn_wmma_f32_16x16x32_bf16(false, qa[0], false, b0, (short)0, z, false, false);
            z = __builtin_amdgcn_wmma_f32_16x16x32_bf16(false, qa[1], false, b1, (short)0, z, false, false);
            sc[t] = z;
        }

        const int mk0 = mrow[key0 + l15];
        const int mk1 = mrow[key0 + 16 + l15];

        // ---- online softmax (row M = r + 8*hi lives across the 16-lane half) ----
        float p0[8], p1[8], scale[8];
#pragma unroll
        for (int r = 0; r < 8; ++r) {
            float s0 = mk0 ? sc[0][r] : -3.0e38f;
            float s1 = mk1 ? sc[1][r] : -3.0e38f;
            float v  = fmaxf(s0, s1);
            v = fmaxf(v, __shfl_xor(v, 1, 32));
            v = fmaxf(v, __shfl_xor(v, 2, 32));
            v = fmaxf(v, __shfl_xor(v, 4, 32));
            v = fmaxf(v, __shfl_xor(v, 8, 32));
            const float mNew = fmaxf(mOld[r], v);
            const float scl  = __expf(mOld[r] - mNew);
            p0[r] = mk0 ? __expf(sc[0][r] - mNew) : 0.0f;
            p1[r] = mk1 ? __expf(sc[1][r] - mNew) : 0.0f;
            float ps = p0[r] + p1[r];
            ps += __shfl_xor(ps, 1, 32);
            ps += __shfl_xor(ps, 2, 32);
            ps += __shfl_xor(ps, 4, 32);
            ps += __shfl_xor(ps, 8, 32);
            lSum[r]  = lSum[r] * scl + ps;
            scale[r] = scl;
            mOld[r]  = mNew;
        }
#pragma unroll
        for (int c = 0; c < 4; ++c)
#pragma unroll
            for (int r = 0; r < 8; ++r) acc[c][r] *= scale[r];

        // ---- P: C-layout -> A-layout via per-wave LDS bounce ----
#pragma unroll
        for (int r = 0; r < 8; ++r) {
            pT[wave][r + 8 * hi][l15]      = to_bf16(p0[r]);
            pT[wave][r + 8 * hi][l15 + 16] = to_bf16(p1[r]);
        }
        asm volatile("s_wait_dscnt 0" ::: "memory");   // same-wave DS in-order
        v16bf pa;
        {
            const __bf16* pp = &pT[wave][l15][hi * 8];
#pragma unroll
            for (int j = 0; j < 8; ++j) { pa[j] = pp[j]; pa[8 + j] = pp[16 + j]; }
        }
        asm volatile("" ::: "memory");

        // ---- O += P(16x32) x V(32x64): 4 WMMAs over d-chunks ----
#pragma unroll
        for (int c = 0; c < 4; ++c) {
            v16bf vb;
            const __bf16* vp = &Vtile[c * 16 + l15][hi * 8];
#pragma unroll
            for (int j = 0; j < 8; ++j) { vb[j] = vp[j]; vb[8 + j] = vp[16 + j]; }
            acc[c] = __builtin_amdgcn_wmma_f32_16x16x32_bf16(false, pa, false, vb, (short)0, acc[c], false, false);
        }

        __syncthreads();   // protect Kt/Vtile before next iteration's staging
    }

    // ---- epilogue: normalize and store fp32 [B,S,HID] ----
#pragma unroll
    for (int c = 0; c < 4; ++c) {
#pragma unroll
        for (int r = 0; r < 8; ++r) {
            const int   srow  = q0 + r + 8 * hi;
            const float denom = lSum[r];
            const float o     = (denom > 0.0f) ? acc[c][r] / denom : 0.0f;
            out[((size_t)bidx * S_ + srow) * HID_ + h * HD_ + c * 16 + l15] = o;
        }
    }
}

extern "C" void kernel_launch(void* const* d_in, const int* in_sizes, int n_in,
                              void* d_out, int out_size, void* d_ws, size_t ws_size,
                              hipStream_t stream) {
    (void)in_sizes; (void)n_in; (void)out_size; (void)ws_size;
    const float* key   = (const float*)d_in[0];
    const float* value = (const float*)d_in[1];
    const float* query = (const float*)d_in[2];
    const int*   amask = (const int*)d_in[3];
    const float* Wk    = (const float*)d_in[4];
    const float* bk    = (const float*)d_in[5];
    const float* Wv    = (const float*)d_in[6];
    const float* bv    = (const float*)d_in[7];
    float* out = (float*)d_out;

    __bf16* Qws = (__bf16*)d_ws;                              // 16 MB
    __bf16* Kws = Qws + (size_t)B_ * NH_ * S_ * HD_;          // 16 MB
    __bf16* Vt  = Kws + (size_t)B_ * NH_ * S_ * HD_;          // 16 MB (total 48 MB)

    proj_kernel<<<dim3(1024, 1, 3), 256, 0, stream>>>(query, key, value,
                                                      Wk, bk, Wv, bv, Qws, Kws, Vt);
    attn_kernel<<<dim3(8, 128), 256, 0, stream>>>(Qws, Kws, Vt, amask, out);
}
